// BinaryMemoryRNN_35553739276666
// MI455X (gfx1250) — matrix operand order repaired and checked by reading, not compiled, not run on hardware
//
#include <hip/hip_runtime.h>

#define BATCH    16384
#define HIDDEN   1024
#define INPUT    1024
#define NUM_BITS 10
#define MEM_SIZE 1024
#define LN_EPS   1e-5f

typedef __attribute__((ext_vector_type(16))) __bf16 v16bf;
typedef __attribute__((ext_vector_type(8)))  __bf16 v8bf;
typedef __attribute__((ext_vector_type(8)))  float  v8f;

// ---------------------------------------------------------------------------
// Kernel 1: fp32 address path. One wave32 per row: logits = h_prev @ M_w^T + M_b,
// hard bits (logit>0), pack to 10-bit index. Pure fp32 so threshold decisions
// match the reference.
// ---------------------------------------------------------------------------
__global__ __launch_bounds__(256) void idx_kernel(
    const float* __restrict__ h_prev, const float* __restrict__ M_w,
    const float* __restrict__ M_b, int* __restrict__ index) {
  const int wave = threadIdx.x >> 5;
  const int lane = threadIdx.x & 31;
  const int row  = blockIdx.x * 8 + wave;
  if (row >= BATCH) return;
  const float* h = h_prev + (size_t)row * HIDDEN;
  float acc[NUM_BITS];
#pragma unroll
  for (int b = 0; b < NUM_BITS; ++b) acc[b] = 0.f;
  for (int k = lane; k < HIDDEN; k += 32) {
    const float hv = h[k];
#pragma unroll
    for (int b = 0; b < NUM_BITS; ++b)
      acc[b] = fmaf(hv, M_w[b * HIDDEN + k], acc[b]);
  }
#pragma unroll
  for (int b = 0; b < NUM_BITS; ++b) {
#pragma unroll
    for (int off = 16; off > 0; off >>= 1)
      acc[b] += __shfl_xor(acc[b], off, 32);
  }
  if (lane == 0) {
    int idx = 0;
#pragma unroll
    for (int b = 0; b < NUM_BITS; ++b)
      if (acc[b] + M_b[b] > 0.f) idx |= 1 << (NUM_BITS - 1 - b);
    index[row] = idx;   // 10 bits -> already in [0, 1023]
  }
}

// ---------------------------------------------------------------------------
// Kernel 2: fused triple GEMM: pre = x@W^T + h_prev@U^T + memory[idx]@Q^T + b.
// Three straight-line K-loops (no in-loop source branching), bf16 WMMA with
// f32 accumulation, register prefetch pipeline over LDS double-barrier.
// Block tile 128x128, BK=32, 8 waves; wave tile 64x32 = 4x2 WMMA tiles.
// ---------------------------------------------------------------------------
#define BM 128
#define BN 128
#define BK 32
#define LDSTRIDE 40   // 32 bf16 + 8 pad per row -> 80B row stride (16B aligned)

__device__ __forceinline__ void load_tile(float4 ga[2][2], float4 gb[2][2],
                                          const float* aP0, const float* aP1,
                                          const float* bP0, const float* bP1,
                                          int kin) {
  ga[0][0] = *(const float4*)(aP0 + kin);
  ga[0][1] = *(const float4*)(aP0 + kin + 4);
  ga[1][0] = *(const float4*)(aP1 + kin);
  ga[1][1] = *(const float4*)(aP1 + kin + 4);
  gb[0][0] = *(const float4*)(bP0 + kin);
  gb[0][1] = *(const float4*)(bP0 + kin + 4);
  gb[1][0] = *(const float4*)(bP1 + kin);
  gb[1][1] = *(const float4*)(bP1 + kin + 4);
}

__device__ __forceinline__ void stage_tile(const float4 ga[2][2], const float4 gb[2][2],
                                           __bf16* sA, __bf16* sB,
                                           int row0, int kq) {
#pragma unroll
  for (int i = 0; i < 2; ++i) {
    v8bf pa, pb;
    pa[0] = (__bf16)ga[i][0].x; pa[1] = (__bf16)ga[i][0].y;
    pa[2] = (__bf16)ga[i][0].z; pa[3] = (__bf16)ga[i][0].w;
    pa[4] = (__bf16)ga[i][1].x; pa[5] = (__bf16)ga[i][1].y;
    pa[6] = (__bf16)ga[i][1].z; pa[7] = (__bf16)ga[i][1].w;
    pb[0] = (__bf16)gb[i][0].x; pb[1] = (__bf16)gb[i][0].y;
    pb[2] = (__bf16)gb[i][0].z; pb[3] = (__bf16)gb[i][0].w;
    pb[4] = (__bf16)gb[i][1].x; pb[5] = (__bf16)gb[i][1].y;
    pb[6] = (__bf16)gb[i][1].z; pb[7] = (__bf16)gb[i][1].w;
    const int row = row0 + i * 64;
    *(v8bf*)&sA[row * LDSTRIDE + kq] = pa;
    *(v8bf*)&sB[row * LDSTRIDE + kq] = pb;
  }
}

__device__ __forceinline__ void mma_step(v8f acc[4][2], const __bf16* sA, const __bf16* sB,
                                         int waveM, int waveN, int half, int l16) {
  // A 16x32 bf16 layout: lane row = lane&15; lanes 0-15 hold K 0-7 & 16-23,
  // lanes 16-31 hold K 8-15 & 24-31.
  v16bf af[4];
#pragma unroll
  for (int mt = 0; mt < 4; ++mt) {
    const int rowA = waveM * 64 + mt * 16 + l16;
    const int kb   = half ? 8 : 0;
    const v8bf lo = *(const v8bf*)&sA[rowA * LDSTRIDE + kb];
    const v8bf hi = *(const v8bf*)&sA[rowA * LDSTRIDE + kb + 16];
    af[mt] = __builtin_shufflevector(lo, hi, 0,1,2,3,4,5,6,7,8,9,10,11,12,13,14,15);
  }
  // B 32x16 bf16 layout: lane col = lane&15; lanes 0-15 hold K 0-7 & 16-23,
  // lanes 16-31 hold K 8-15 & 24-31.
  v16bf bfrag[2];
#pragma unroll
  for (int nt = 0; nt < 2; ++nt) {
    const int rowB = waveN * 32 + nt * 16 + l16;
    const int kb   = half ? 8 : 0;
    const v8bf lo = *(const v8bf*)&sB[rowB * LDSTRIDE + kb];
    const v8bf hi = *(const v8bf*)&sB[rowB * LDSTRIDE + kb + 16];
    bfrag[nt] = __builtin_shufflevector(lo, hi, 0,1,2,3,4,5,6,7,8,9,10,11,12,13,14,15);
  }
#pragma unroll
  for (int mt = 0; mt < 4; ++mt)
#pragma unroll
    for (int nt = 0; nt < 2; ++nt)
      acc[mt][nt] = __builtin_amdgcn_wmma_f32_16x16x32_bf16(
          false, af[mt], false, bfrag[nt], (short)0, acc[mt][nt], false, false);
}

__global__ __launch_bounds__(256) void gemm_kernel(
    const float* __restrict__ x, const float* __restrict__ h_prev,
    const float* __restrict__ memory, const int* __restrict__ index,
    const float* __restrict__ W_w, const float* __restrict__ U_w,
    const float* __restrict__ Q_w, const float* __restrict__ W_b,
    const float* __restrict__ U_b, const float* __restrict__ Q_b,
    float* __restrict__ pre) {
  __shared__ __bf16 sA[BM * LDSTRIDE];
  __shared__ __bf16 sB[BN * LDSTRIDE];
  __shared__ int    sIdx[BM];

  const int tid     = threadIdx.x;
  const int lane    = tid & 31;
  const int waveId  = tid >> 5;
  const int waveM   = waveId & 1;   // 2 row bands of 64
  const int waveN   = waveId >> 1;  // 4 col bands of 32
  const int half    = lane >> 4;
  const int l16     = lane & 15;
  const int blockN0 = blockIdx.x * BN;
  const int blockM0 = blockIdx.y * BM;

  // Per-thread staging slots: rows {row0, row0+64}, 8 contiguous k at kq.
  const int row0 = tid >> 2;
  const int kq   = (tid & 3) * 8;

  if (tid < BM) sIdx[tid] = index[blockM0 + tid];
  __syncthreads();

  v8f acc[4][2];
#pragma unroll
  for (int mt = 0; mt < 4; ++mt)
#pragma unroll
    for (int nt = 0; nt < 2; ++nt) acc[mt][nt] = (v8f)(0.f);

  // Hoisted per-thread source pointers (gather resolved once for segment 2).
  const size_t aOff0 = (size_t)(blockM0 + row0) * 1024 + kq;
  const size_t aOff1 = (size_t)(blockM0 + row0 + 64) * 1024 + kq;
  const size_t bOff0 = (size_t)(blockN0 + row0) * 1024 + kq;
  const size_t bOff1 = (size_t)(blockN0 + row0 + 64) * 1024 + kq;
  const float* aSeg0[3] = { x + aOff0, h_prev + aOff0,
                            memory + (size_t)sIdx[row0] * 1024 + kq };
  const float* aSeg1[3] = { x + aOff1, h_prev + aOff1,
                            memory + (size_t)sIdx[row0 + 64] * 1024 + kq };
  const float* bSeg0[3] = { W_w + bOff0, U_w + bOff0, Q_w + bOff0 };
  const float* bSeg1[3] = { W_w + bOff1, U_w + bOff1, Q_w + bOff1 };

  float4 ga[2][2], gb[2][2];
#pragma unroll
  for (int seg = 0; seg < 3; ++seg) {
    const float* aP0 = aSeg0[seg];
    const float* aP1 = aSeg1[seg];
    const float* bP0 = bSeg0[seg];
    const float* bP1 = bSeg1[seg];
    load_tile(ga, gb, aP0, aP1, bP0, bP1, 0);
    for (int s = 0; s < 32; ++s) {
      stage_tile(ga, gb, sA, sB, row0, kq);
      __syncthreads();
      // Branchless prefetch of next step (wraps to 0 on the last; harmless).
      load_tile(ga, gb, aP0, aP1, bP0, bP1, ((s + 1) & 31) * BK);
      mma_step(acc, sA, sB, waveM, waveN, half, l16);
      __syncthreads();
    }
  }

  // Epilogue: add summed biases, store fp32 pre-activations.
  // C/D layout: VGPR r -> M = r (+8 for lanes 16-31), N = lane&15.
#pragma unroll
  for (int nt = 0; nt < 2; ++nt) {
    const int n    = blockN0 + waveN * 32 + nt * 16 + l16;
    const float bb = W_b[n] + U_b[n] + Q_b[n];
#pragma unroll
    for (int mt = 0; mt < 4; ++mt) {
      const int m0 = blockM0 + waveM * 64 + mt * 16 + half * 8;
#pragma unroll
      for (int r = 0; r < 8; ++r)
        pre[(size_t)(m0 + r) * HIDDEN + n] = acc[mt][nt][r] + bb;
    }
  }
}

// ---------------------------------------------------------------------------
// Kernel 3: LayerNorm (biased var) + affine + sigmoid. One 256-thread block
// per row; each thread owns 4 contiguous elements.
// ---------------------------------------------------------------------------
__global__ __launch_bounds__(256) void ln_kernel(
    const float* __restrict__ pre, const float* __restrict__ ln_g,
    const float* __restrict__ ln_b, float* __restrict__ out) {
  __shared__ float redS[8], redS2[8];
  const int row  = blockIdx.x;
  const int tid  = threadIdx.x;
  const int lane = tid & 31;
  const int wv   = tid >> 5;
  const float* p = pre + (size_t)row * HIDDEN;
  const int k0   = tid * 4;
  const float4 v = *(const float4*)(p + k0);
  float s  = v.x + v.y + v.z + v.w;
  float s2 = v.x * v.x + v.y * v.y + v.z * v.z + v.w * v.w;
#pragma unroll
  for (int off = 16; off > 0; off >>= 1) {
    s  += __shfl_xor(s,  off, 32);
    s2 += __shfl_xor(s2, off, 32);
  }
  if (lane == 0) { redS[wv] = s; redS2[wv] = s2; }
  __syncthreads();
  float ts = 0.f, ts2 = 0.f;
#pragma unroll
  for (int i = 0; i < 8; ++i) { ts += redS[i]; ts2 += redS2[i]; }
  const float mu   = ts * (1.f / HIDDEN);
  const float var  = ts2 * (1.f / HIDDEN) - mu * mu;
  const float rstd = rsqrtf(var + LN_EPS);
  const float e[4] = {v.x, v.y, v.z, v.w};
  float4 o;
  float* op = &o.x;
#pragma unroll
  for (int i = 0; i < 4; ++i) {
    const float y = (e[i] - mu) * rstd * ln_g[k0 + i] + ln_b[k0 + i];
    op[i] = 1.f / (1.f + __expf(-y));
  }
  *(float4*)(out + (size_t)row * HIDDEN + k0) = o;
}

// ---------------------------------------------------------------------------
extern "C" void kernel_launch(void* const* d_in, const int* in_sizes, int n_in,
                              void* d_out, int out_size, void* d_ws, size_t ws_size,
                              hipStream_t stream) {
  (void)in_sizes; (void)n_in; (void)out_size; (void)ws_size;
  const float* x      = (const float*)d_in[0];
  const float* h_prev = (const float*)d_in[1];
  const float* memory = (const float*)d_in[2];
  const float* W_w    = (const float*)d_in[3];
  const float* W_b    = (const float*)d_in[4];
  const float* U_w    = (const float*)d_in[5];
  const float* U_b    = (const float*)d_in[6];
  const float* Q_w    = (const float*)d_in[7];
  const float* Q_b    = (const float*)d_in[8];
  const float* M_w    = (const float*)d_in[9];
  const float* M_b    = (const float*)d_in[10];
  const float* ln_g   = (const float*)d_in[11];
  const float* ln_b   = (const float*)d_in[12];
  float* out = (float*)d_out;

  int*   idx = (int*)d_ws;                                   // 64 KB
  float* pre = (float*)((char*)d_ws + 256 * 1024);           // 64 MB scratch

  idx_kernel<<<BATCH / 8, 256, 0, stream>>>(h_prev, M_w, M_b, idx);
  dim3 grid(HIDDEN / BN, BATCH / BM);   // (8, 128)
  gemm_kernel<<<grid, 256, 0, stream>>>(x, h_prev, memory, idx,
                                        W_w, U_w, Q_w, W_b, U_b, Q_b, pre);
  ln_kernel<<<BATCH, 256, 0, stream>>>(pre, ln_g, ln_b, out);
}
